// GNN_v2_51049981280776
// MI455X (gfx1250) — compile-verified, matrix operand order checked by reading, not run on hardware
//
#include <hip/hip_runtime.h>
#include <hip/hip_bf16.h>
#include <math.h>

typedef __attribute__((ext_vector_type(16))) __bf16 v16bf;
typedef __attribute__((ext_vector_type(8)))  float  v8f;

#define N_NODES  10000
#define N_EDGES  160000
#define E_TOT    (N_EDGES + N_NODES)   // with self loops
#define HEADS    4
#define CDIM     128
#define HC       512                   // HEADS * CDIM
#define X_COLS   384
#define FEAT_IN  256
#define TOPO_IN  128
#define N_GRAPHS 8
#define FC_IN    1024
#define OUT_DIM  128
#define NEG_SLOPE 0.2f

// ---------------- utility kernels ----------------

__global__ __launch_bounds__(256) void k_fill(float* p, float v, int n) {
    int i = blockIdx.x * blockDim.x + threadIdx.x;
    if (i < n) p[i] = v;
}

// strided f32 -> bf16 convert (handles x column slices and full buffers)
__global__ __launch_bounds__(256) void k_cvt_bf16(const float* __restrict__ in, int ld, int cols,
                                                  __bf16* __restrict__ out, int n) {
    int i = blockIdx.x * blockDim.x + threadIdx.x;
    if (i >= n) return;
    int r = i / cols, c = i - r * cols;
    out[i] = (__bf16)in[(size_t)r * ld + c];
}

// W[K x Nout] f32 -> Wt[Nout x K] bf16 (row-major in the transposed view)
__global__ __launch_bounds__(256) void k_wt_bf16(const float* __restrict__ W,
                                                 __bf16* __restrict__ Wt, int K, int Nout) {
    int i = blockIdx.x * blockDim.x + threadIdx.x;
    if (i >= K * Nout) return;
    int nn = i / K, k = i - nn * K;
    Wt[i] = (__bf16)W[(size_t)k * Nout + nn];
}

// ---------------- WMMA GEMM: C[M,Nout] = A[M,K](bf16) @ Bt[Nout,K]^T, f32 out ----------------
// One wave -> 16(M) x 32(N) output strip, 2 WMMAs per K-step reusing the A fragment.
__global__ __launch_bounds__(256) void k_gemm_wmma(const __bf16* __restrict__ A,
                                                   const __bf16* __restrict__ Bt,
                                                   float* __restrict__ C,
                                                   int M, int K, int Nout) {
    const int lane = threadIdx.x & 31;
    const int wave = (blockIdx.x * blockDim.x + threadIdx.x) >> 5;
    const int nT = Nout >> 5;                 // 32-wide N tiles
    const int mT = M >> 4;                    // 16-wide M tiles
    if (wave >= mT * nT) return;              // uniform per wave
    const int mt = wave / nT;
    const int nt = wave - mt * nT;

    // A fragment: lane L -> row = mt*16 + (L&15), K half = (L>>4)*16, 16 contiguous bf16
    const int arow = (mt << 4) + (lane & 15);
    const int koff = (lane >> 4) << 4;
    // B fragments: lane L -> col n, 16 contiguous K bf16 from the transposed weights
    const int n0 = (nt << 5) + (lane & 15);
    const int n1 = n0 + 16;

    const __bf16* aP  = A  + (size_t)arow * K + koff;
    const __bf16* b0P = Bt + (size_t)n0   * K + koff;
    const __bf16* b1P = Bt + (size_t)n1   * K + koff;

    v8f acc0 = {}; v8f acc1 = {};
    for (int k = 0; k < K; k += 32) {
        v16bf af = *(const v16bf*)(aP  + k);
        v16bf b0 = *(const v16bf*)(b0P + k);
        v16bf b1 = *(const v16bf*)(b1P + k);
        __builtin_prefetch(aP + k + 32, 0, 1);   // global_prefetch_b8 for next K-step
        acc0 = __builtin_amdgcn_wmma_f32_16x16x32_bf16(false, af, false, b0,
                                                       (short)0, acc0, false, false);
        acc1 = __builtin_amdgcn_wmma_f32_16x16x32_bf16(false, af, false, b1,
                                                       (short)0, acc1, false, false);
    }

    // C/D layout: VGPR v, lanes 0-15 -> M = v, lanes 16-31 -> M = 8+v; N = lane&15
    const int cn0   = (nt << 5) + (lane & 15);
    const int rbase = (mt << 4) + ((lane >> 4) << 3);
    for (int v = 0; v < 8; ++v) {
        C[(size_t)(rbase + v) * Nout + cn0]      = acc0[v];
        C[(size_t)(rbase + v) * Nout + cn0 + 16] = acc1[v];
    }
}

// ---------------- attention scores: s_src/s_dst [N, H] ----------------
__global__ __launch_bounds__(256) void k_attn(const float* __restrict__ h,
                                              const float* __restrict__ a_src,
                                              const float* __restrict__ a_dst,
                                              float* __restrict__ s_src,
                                              float* __restrict__ s_dst) {
    int i = blockIdx.x * blockDim.x + threadIdx.x;
    if (i >= N_NODES * HEADS) return;
    int n = i / HEADS, hh = i - n * HEADS;
    const float* hp = h + (size_t)n * HC + hh * CDIM;
    const float* as = a_src + hh * CDIM;
    const float* ad = a_dst + hh * CDIM;
    float ss = 0.f, sd = 0.f;
    for (int c = 0; c < CDIM; ++c) { float v = hp[c]; ss += v * as[c]; sd += v * ad[c]; }
    s_src[i] = ss; s_dst[i] = sd;
}

__device__ inline void atomicMaxF(float* addr, float val) {
    // valid with -inf init: positives via signed int max, negatives via unsigned min
    if (val >= 0.f) atomicMax((int*)addr, __float_as_int(val));
    else            atomicMin((unsigned int*)addr, (unsigned int)__float_as_int(val));
}

// ---------------- edge phase ----------------
__global__ __launch_bounds__(256) void k_edge_logit(const int* __restrict__ ei,
                                                    const float* __restrict__ s_src,
                                                    const float* __restrict__ s_dst,
                                                    float* __restrict__ ebuf,
                                                    float* __restrict__ emax) {
    int i = blockIdx.x * blockDim.x + threadIdx.x;
    if (i >= E_TOT * HEADS) return;
    int e = i / HEADS, hh = i - e * HEADS;
    int s, d;
    if (e < N_EDGES) { s = ei[e]; d = ei[N_EDGES + e]; } else { s = d = e - N_EDGES; }
    float v = s_src[s * HEADS + hh] + s_dst[d * HEADS + hh];
    v = v > 0.f ? v : NEG_SLOPE * v;
    ebuf[i] = v;
    atomicMaxF(&emax[d * HEADS + hh], v);
}

__global__ __launch_bounds__(256) void k_fix_nonfinite(float* p, int n) {
    int i = blockIdx.x * blockDim.x + threadIdx.x;
    if (i < n) { float v = p[i]; if (!isfinite(v)) p[i] = 0.f; }
}

__global__ __launch_bounds__(256) void k_edge_exp(const int* __restrict__ ei,
                                                  float* __restrict__ ebuf,
                                                  const float* __restrict__ emax,
                                                  float* __restrict__ denom) {
    int i = blockIdx.x * blockDim.x + threadIdx.x;
    if (i >= E_TOT * HEADS) return;
    int e = i / HEADS, hh = i - e * HEADS;
    int d = (e < N_EDGES) ? ei[N_EDGES + e] : (e - N_EDGES);
    float ex = __expf(ebuf[i] - emax[d * HEADS + hh]);
    ebuf[i] = ex;
    atomicAdd(&denom[d * HEADS + hh], ex);
}

// thread per (edge, feature): out[dst, r] += h[src, r] * alpha[e, head(r)]
__global__ __launch_bounds__(256) void k_aggregate(const int* __restrict__ ei,
                                                   const float* __restrict__ h,
                                                   const float* __restrict__ ebuf,
                                                   const float* __restrict__ denom,
                                                   float* __restrict__ out) {
    long long i = (long long)blockIdx.x * blockDim.x + threadIdx.x;
    if (i >= (long long)E_TOT * HC) return;
    int e = (int)(i >> 9);      // / 512
    int r = (int)(i & 511);
    int hh = r >> 7;            // / 128
    int s, d;
    if (e < N_EDGES) { s = ei[e]; d = ei[N_EDGES + e]; } else { s = d = e - N_EDGES; }
    float alpha = ebuf[e * HEADS + hh] / (denom[d * HEADS + hh] + 1e-16f);
    atomicAdd(&out[(size_t)d * HC + r], h[(size_t)s * HC + r] * alpha);
}

__global__ __launch_bounds__(256) void k_bias_relu(float* __restrict__ out,
                                                   const float* __restrict__ b, int n) {
    int i = blockIdx.x * blockDim.x + threadIdx.x;
    if (i >= n) return;
    float v = out[i] + b[i & (HC - 1)];
    out[i] = v > 0.f ? v : 0.f;
}

// ---------------- pooling + FC ----------------
__global__ __launch_bounds__(256) void k_pool(const float* __restrict__ pf,
                                              const float* __restrict__ pt,
                                              const int* __restrict__ batch,
                                              float* __restrict__ pooled) {
    int i = blockIdx.x * blockDim.x + threadIdx.x;
    if (i >= N_NODES * FC_IN) return;
    int n = i >> 10, j = i & 1023;
    float v = (j < HC) ? pf[(size_t)n * HC + j] : pt[(size_t)n * HC + (j - HC)];
    atomicMaxF(&pooled[batch[n] * FC_IN + j], v);
}

__global__ __launch_bounds__(256) void k_fc_sigmoid(const float* __restrict__ pooled,
                                                    const float* __restrict__ W,
                                                    const float* __restrict__ b,
                                                    float* __restrict__ out) {
    int i = blockIdx.x * blockDim.x + threadIdx.x;
    if (i >= N_GRAPHS * OUT_DIM) return;
    int g = i / OUT_DIM, o = i - g * OUT_DIM;
    float acc = b[o];
    for (int j = 0; j < FC_IN; ++j) {
        float p = pooled[g * FC_IN + j];
        if (!isfinite(p)) p = 0.f;
        acc += p * W[(size_t)j * OUT_DIM + o];
    }
    out[i] = 1.f / (1.f + __expf(-acc));
}

// ---------------- host orchestration ----------------

static inline int cdiv(long long a, int b) { return (int)((a + b - 1) / b); }

extern "C" void kernel_launch(void* const* d_in, const int* in_sizes, int n_in,
                              void* d_out, int out_size, void* d_ws, size_t ws_size,
                              hipStream_t stream) {
    const float* x     = (const float*)d_in[0];
    const int*   ei    = (const int*)d_in[1];
    const int*   batch = (const int*)d_in[2];
    const float* fc_W  = (const float*)d_in[27];
    const float* fc_b  = (const float*)d_in[28];

    char* ws = (char*)d_ws;
    size_t off = 0;
    auto alloc = [&](size_t bytes) -> char* {
        char* p = ws + off;
        off += (bytes + 255) & ~(size_t)255;
        return p;
    };
    __bf16* xbf   = (__bf16*)alloc((size_t)N_NODES * HC * sizeof(__bf16));
    __bf16* wt    = (__bf16*)alloc((size_t)HC * HC * sizeof(__bf16));
    float*  hbuf  = (float*)alloc((size_t)N_NODES * HC * sizeof(float));
    float*  bufA  = (float*)alloc((size_t)N_NODES * HC * sizeof(float));
    float*  bufB  = (float*)alloc((size_t)N_NODES * HC * sizeof(float));
    float*  pf    = (float*)alloc((size_t)N_NODES * HC * sizeof(float));
    float*  pt    = (float*)alloc((size_t)N_NODES * HC * sizeof(float));
    float*  ssrc  = (float*)alloc((size_t)N_NODES * HEADS * sizeof(float));
    float*  sdst  = (float*)alloc((size_t)N_NODES * HEADS * sizeof(float));
    float*  emax  = (float*)alloc((size_t)N_NODES * HEADS * sizeof(float));
    float*  denom = (float*)alloc((size_t)N_NODES * HEADS * sizeof(float));
    float*  ebuf  = (float*)alloc((size_t)E_TOT * HEADS * sizeof(float));
    float*  pooled= (float*)alloc((size_t)N_GRAPHS * FC_IN * sizeof(float));

    const float NEG_INF = -__builtin_inff();

    for (int br = 0; br < 2; ++br) {
        const float* cur = x + (br == 0 ? 0 : FEAT_IN);
        int cur_ld   = X_COLS;
        int cur_cols = (br == 0) ? FEAT_IN : TOPO_IN;

        for (int l = 0; l < 3; ++l) {
            int pbase = 3 + br * 12 + l * 4;
            const float* W     = (const float*)d_in[pbase + 0];
            const float* a_src = (const float*)d_in[pbase + 1];
            const float* a_dst = (const float*)d_in[pbase + 2];
            const float* bias  = (const float*)d_in[pbase + 3];
            const int K = cur_cols;
            float* outb = (l == 0) ? bufA : (l == 1) ? bufB : (br == 0 ? pf : pt);

            // 1. convert activations to bf16
            int ncv = N_NODES * K;
            k_cvt_bf16<<<cdiv(ncv, 256), 256, 0, stream>>>(cur, cur_ld, K, xbf, ncv);
            // 2. transpose + convert weights to [HC x K] bf16
            k_wt_bf16<<<cdiv(K * HC, 256), 256, 0, stream>>>(W, wt, K, HC);
            // 3. WMMA projection h = x @ W  (625 M-tiles x 16 N-tiles = 10000 wave tasks)
            int waves = (N_NODES / 16) * (HC / 32);
            k_gemm_wmma<<<cdiv((long long)waves * 32, 256), 256, 0, stream>>>(
                xbf, wt, hbuf, N_NODES, K, HC);
            // 4. attention scores
            k_attn<<<cdiv(N_NODES * HEADS, 256), 256, 0, stream>>>(hbuf, a_src, a_dst, ssrc, sdst);
            // 5. init reduction buffers
            k_fill<<<cdiv(N_NODES * HEADS, 256), 256, 0, stream>>>(emax, NEG_INF, N_NODES * HEADS);
            k_fill<<<cdiv(N_NODES * HEADS, 256), 256, 0, stream>>>(denom, 0.f, N_NODES * HEADS);
            k_fill<<<cdiv(N_NODES * HC, 256), 256, 0, stream>>>(outb, 0.f, N_NODES * HC);
            // 6. edge softmax + aggregation
            k_edge_logit<<<cdiv(E_TOT * HEADS, 256), 256, 0, stream>>>(ei, ssrc, sdst, ebuf, emax);
            k_fix_nonfinite<<<cdiv(N_NODES * HEADS, 256), 256, 0, stream>>>(emax, N_NODES * HEADS);
            k_edge_exp<<<cdiv(E_TOT * HEADS, 256), 256, 0, stream>>>(ei, ebuf, emax, denom);
            k_aggregate<<<cdiv((long long)E_TOT * HC, 256), 256, 0, stream>>>(
                ei, hbuf, ebuf, denom, outb);
            // 7. bias + relu epilogue
            k_bias_relu<<<cdiv(N_NODES * HC, 256), 256, 0, stream>>>(outb, bias, N_NODES * HC);

            cur = outb; cur_ld = HC; cur_cols = HC;
        }
    }

    // global max-pool over batch segments, then FC + sigmoid
    k_fill<<<cdiv(N_GRAPHS * FC_IN, 256), 256, 0, stream>>>(pooled, NEG_INF, N_GRAPHS * FC_IN);
    k_pool<<<cdiv(N_NODES * FC_IN, 256), 256, 0, stream>>>(pf, pt, batch, pooled);
    k_fc_sigmoid<<<cdiv(N_GRAPHS * OUT_DIM, 256), 256, 0, stream>>>(
        pooled, fc_W, fc_b, (float*)d_out);
}
